// MultiHeadGATLayer_64338610094367
// MI455X (gfx1250) — matrix operand order repaired and checked by reading, not compiled, not run on hardware
//
#include <hip/hip_runtime.h>
#include <hip/hip_bf16.h>
#include <math.h>

#define N_SRC   10000
#define N_TGT   10000
#define NEDGE   160000
#define IN_DIM  512
#define N_HEAD  8
#define OUT_DIM 64
#define DMODEL  512          // N_HEAD*OUT_DIM
#define EDGE_DIM 128
#define FFN_HID 2048
#define LN_EPS  1e-5f

typedef __bf16 v16bf __attribute__((ext_vector_type(16)));
typedef float  v8f   __attribute__((ext_vector_type(8)));
typedef unsigned int v4u __attribute__((ext_vector_type(4)));

union FragU { v4u q[2]; v16bf f; };

__device__ __forceinline__ float warp_sum(float v) {
#pragma unroll
    for (int off = 16; off > 0; off >>= 1) v += __shfl_xor(v, off, 32);
    return v;
}

__device__ __forceinline__ void atomicMaxFloat(float* addr, float val) {
    if (val >= 0.0f) atomicMax((int*)addr, __float_as_int(val));
    else             atomicMin((unsigned int*)addr, __float_as_uint(val));
}

__device__ __forceinline__ unsigned int pack_bf16(float lo, float hi) {
    unsigned short ul = __builtin_bit_cast(unsigned short, (__bf16)lo);
    unsigned short uh = __builtin_bit_cast(unsigned short, (__bf16)hi);
    return (unsigned int)ul | ((unsigned int)uh << 16);
}

__device__ __forceinline__ v16bf load_frag2(const unsigned int* row, int o0, int o1) {
    FragU t;
    t.q[0] = *(const v4u*)(row + o0);
    t.q[1] = *(const v4u*)(row + o1);
    return t.f;
}

// ---------------------------------------------------------------------------
// bf16 WMMA GEMM:  C[M,N] = epi(A[M,K] @ Bt[N,K]^T)
//   A  : bf16 row-major [M][K]
//   Bt : bf16 pre-transposed [N][K]
// MODE 0: plain   MODE 1: relu(x+bias)   MODE 2: x+bias+residual
// OUTBF: store output as bf16 (else fp32)
// 256 threads = 8 waves; block tile 128x64; wave tile 32x32 (4 wmma accums).
// Double-buffered LDS; staging = pure aligned b128 copies (no conversion).
// ---------------------------------------------------------------------------
#define BM 128
#define BN 64
#define BK 32
#define LSTRIDE 20   // uints per LDS row: 16 data pairs + 4 pad (80B, 16B aligned)

struct StageRegs { v4u a0, a1, b; bool m0ok, m1ok; };

__device__ __forceinline__ void stage_load(const __bf16* __restrict__ A,
                                           const __bf16* __restrict__ Bt,
                                           int M, int K, int m0, int n0, int k0,
                                           int tid, StageRegs& s)
{
    // A tile 128x32 bf16 = 8KB: 2 x 16B per thread
    {
        int q = tid;                  // chunk 0
        int r = q >> 2, c16 = q & 3;
        int gm = m0 + r;
        int gmc = gm < M ? gm : M - 1;
        s.m0ok = gm < M;
        s.a0 = *(const v4u*)(A + (long)gmc * K + k0 + c16 * 8);
    }
    {
        int q = tid + 256;            // chunk 1
        int r = q >> 2, c16 = q & 3;
        int gm = m0 + r;
        int gmc = gm < M ? gm : M - 1;
        s.m1ok = gm < M;
        s.a1 = *(const v4u*)(A + (long)gmc * K + k0 + c16 * 8);
    }
    // B tile 64x32 bf16 = 4KB: 1 x 16B per thread (Bt row-contiguous)
    {
        int r = tid >> 2, c16 = tid & 3;
        s.b = *(const v4u*)(Bt + (long)(n0 + r) * K + k0 + c16 * 8);
    }
}

__device__ __forceinline__ void stage_store(unsigned int (*AsU)[LSTRIDE],
                                            unsigned int (*BsU)[LSTRIDE],
                                            int tid, StageRegs& s)
{
    const v4u vzero = {0u, 0u, 0u, 0u};
    {
        int q = tid;
        int r = q >> 2, c16 = q & 3;
        v4u v = s.a0; if (!s.m0ok) v = vzero;
        *(v4u*)&AsU[r][c16 * 4] = v;
    }
    {
        int q = tid + 256;
        int r = q >> 2, c16 = q & 3;
        v4u v = s.a1; if (!s.m1ok) v = vzero;
        *(v4u*)&AsU[r][c16 * 4] = v;
    }
    {
        int r = tid >> 2, c16 = tid & 3;
        *(v4u*)&BsU[r][c16 * 4] = s.b;
    }
}

template <int MODE, bool OUTBF>
__global__ __launch_bounds__(256)
void gemm_bf16_wmma(const __bf16* __restrict__ A, const __bf16* __restrict__ Bt,
                    void* __restrict__ CoutV, const float* __restrict__ bias,
                    const float* __restrict__ residual, int M, int N, int K)
{
    __shared__ unsigned int AsU[2][BM][LSTRIDE];  // [buf][m][k-pair]
    __shared__ unsigned int BsU[2][BN][LSTRIDE];  // [buf][n][k-pair]

    const int tid  = threadIdx.x;
    const int lane = tid & 31;
    const int wave = tid >> 5;
    const int m0 = blockIdx.x * BM;
    const int n0 = blockIdx.y * BN;
    const int wm = (wave & 3) * 32;   // 0,32,64,96
    const int wn = (wave >> 2) * 32;  // 0,32

    v8f acc00 = {}, acc01 = {}, acc10 = {}, acc11 = {};

    const int ka2   = (lane >> 4) * 4;
    const int kb2   = (lane >> 4) * 8;
    const int arow0 = wm + (lane & 15);
    const int arow1 = arow0 + 16;
    const int bcol  = lane & 15;

    StageRegs sr;
    stage_load(A, Bt, M, K, m0, n0, 0, tid, sr);
    stage_store(AsU[0], BsU[0], tid, sr);
    __syncthreads();

    int cur = 0;
    for (int k0 = 0; k0 < K; k0 += BK) {
        const bool nxt = (k0 + BK) < K;
        if (nxt) stage_load(A, Bt, M, K, m0, n0, k0 + BK, tid, sr);

        v16bf a0 = load_frag2(&AsU[cur][arow0][0], ka2, 8 + ka2);
        v16bf a1 = load_frag2(&AsU[cur][arow1][0], ka2, 8 + ka2);
        v16bf b0 = load_frag2(&BsU[cur][wn + bcol][0],      ka2, 8 + ka2);
        v16bf b1 = load_frag2(&BsU[cur][wn + 16 + bcol][0], ka2, 8 + ka2);

        acc00 = __builtin_amdgcn_wmma_f32_16x16x32_bf16(false, a0, false, b0, (short)0, acc00, false, false);
        acc01 = __builtin_amdgcn_wmma_f32_16x16x32_bf16(false, a0, false, b1, (short)0, acc01, false, false);
        acc10 = __builtin_amdgcn_wmma_f32_16x16x32_bf16(false, a1, false, b0, (short)0, acc10, false, false);
        acc11 = __builtin_amdgcn_wmma_f32_16x16x32_bf16(false, a1, false, b1, (short)0, acc11, false, false);

        if (nxt) stage_store(AsU[cur ^ 1], BsU[cur ^ 1], tid, sr);
        __syncthreads();
        cur ^= 1;
    }

    // epilogue; C/D layout: vgpr r -> row (lane>=16?8:0)+r, col lane&15
    float*  cf = (float*)CoutV;
    __bf16* cb = (__bf16*)CoutV;
    const int crow = (lane >> 4) * 8;
    const int ccol = lane & 15;
#pragma unroll
    for (int sub = 0; sub < 2; ++sub) {
        const v8f* accl = sub ? &acc10 : &acc00;
        const v8f* acch = sub ? &acc11 : &acc01;
#pragma unroll
        for (int r = 0; r < 8; ++r) {
            int gm = m0 + wm + sub * 16 + crow + r;
            if (gm < M) {
                int c0 = n0 + wn + ccol;
                int c1 = c0 + 16;
                float v0 = (*accl)[r], v1 = (*acch)[r];
                if (MODE == 1) {
                    v0 = fmaxf(v0 + bias[c0], 0.0f);
                    v1 = fmaxf(v1 + bias[c1], 0.0f);
                } else if (MODE == 2) {
                    v0 = v0 + bias[c0] + residual[(long)gm * N + c0];
                    v1 = v1 + bias[c1] + residual[(long)gm * N + c1];
                }
                if (OUTBF) {
                    cb[(long)gm * N + c0] = (__bf16)v0;
                    cb[(long)gm * N + c1] = (__bf16)v1;
                } else {
                    cf[(long)gm * N + c0] = v0;
                    cf[(long)gm * N + c1] = v1;
                }
            }
        }
    }
}

// ---------------------------------------------------------------------------
// pre-pack: fp32 -> bf16 (same layout), processed as pairs
// ---------------------------------------------------------------------------
__global__ void pack_bf16_kernel(const float* __restrict__ in,
                                 unsigned int* __restrict__ outU, long npairs)
{
    long i = (long)blockIdx.x * blockDim.x + threadIdx.x;
    if (i >= npairs) return;
    float2 v = *(const float2*)&in[i * 2];
    outU[i] = pack_bf16(v.x, v.y);
}

// W[K][N] fp32 -> Wt[N][K] bf16 (paired along K); K,N multiples of 32
__global__ __launch_bounds__(256)
void transpose_pack_kernel(const float* __restrict__ W, unsigned int* __restrict__ WtU,
                           int K, int N)
{
    __shared__ float tile[32][33];
    int k0 = blockIdx.x * 32, n0 = blockIdx.y * 32;
    int tid = threadIdx.x;
#pragma unroll
    for (int i = 0; i < 4; ++i) {
        int q = tid + i * 256;            // 0..1023
        int r = q >> 5, c = q & 31;       // r: k, c: n
        tile[r][c] = W[(long)(k0 + r) * N + n0 + c];
    }
    __syncthreads();
#pragma unroll
    for (int i = 0; i < 2; ++i) {
        int q = tid + i * 256;            // 0..511
        int n = q >> 4, kp = q & 15;
        WtU[(long)(n0 + n) * (K / 2) + (k0 / 2) + kp] =
            pack_bf16(tile[kp * 2][n], tile[kp * 2 + 1][n]);
    }
}

// ---------------------------------------------------------------------------
// C[r][h] = sum_d W_feat[r, h*64+d] * a_feat[h, d]
// ---------------------------------------------------------------------------
__global__ void compute_C_kernel(const float* __restrict__ W_feat,
                                 const float* __restrict__ attn_a,
                                 float* __restrict__ Cmat)
{
    int r = threadIdx.x;  // 128 threads
    const float* wrow = W_feat + (long)r * DMODEL;
#pragma unroll
    for (int h = 0; h < N_HEAD; ++h) {
        const float* af = attn_a + h * 3 * OUT_DIM + 2 * OUT_DIM;
        float s = 0.0f;
        for (int d = 0; d < OUT_DIM; ++d) s += wrow[h * OUT_DIM + d] * af[d];
        Cmat[r * N_HEAD + h] = s;
    }
}

// s_src[n,h] = sum_d z[n, h*64+d] * a_src[h,d]
__global__ __launch_bounds__(256)
void s_src_kernel(const float* __restrict__ z, const float* __restrict__ attn_a,
                  float* __restrict__ s_src)
{
    int row  = blockIdx.x;
    int head = threadIdx.x >> 5;
    int lane = threadIdx.x & 31;
    const float* zr = z + (long)row * DMODEL + head * OUT_DIM;
    const float* as = attn_a + head * 3 * OUT_DIM;
    float p = zr[lane] * as[lane] + zr[lane + 32] * as[lane + 32];
    p = warp_sum(p);
    if (lane == 0) s_src[row * N_HEAD + head] = p;
}

// per-edge score: e = leaky_relu(s_src[src] + edge_embed@C); segment-max
__global__ __launch_bounds__(256)
void edge_score_kernel(const float* __restrict__ edge_embed, const int* __restrict__ edge_src,
                       const int* __restrict__ edge_dst, const float* __restrict__ Cmat,
                       const float* __restrict__ s_src, float* __restrict__ ebuf,
                       float* __restrict__ mmax)
{
    __shared__ float Cs[EDGE_DIM][N_HEAD + 1];
    for (int i = threadIdx.x; i < EDGE_DIM * N_HEAD; i += 256)
        Cs[i >> 3][i & 7] = Cmat[i];
    __syncthreads();

    int wave = threadIdx.x >> 5;
    int lane = threadIdx.x & 31;
    long e = (long)blockIdx.x * 8 + wave;
    if (e >= NEDGE) return;

    const float* ee = edge_embed + e * EDGE_DIM;
    float f[N_HEAD];
#pragma unroll
    for (int h = 0; h < N_HEAD; ++h) f[h] = 0.0f;
#pragma unroll
    for (int j = 0; j < 4; ++j) {
        float v = ee[lane + j * 32];
#pragma unroll
        for (int h = 0; h < N_HEAD; ++h) f[h] += v * Cs[lane + j * 32][h];
    }
#pragma unroll
    for (int h = 0; h < N_HEAD; ++h) f[h] = warp_sum(f[h]);

    if (lane < N_HEAD) {
        int src = edge_src[e], dst = edge_dst[e];
        float v = s_src[src * N_HEAD + lane] + f[lane];
        v = (v > 0.0f) ? v : 0.01f * v;
        ebuf[e * N_HEAD + lane] = v;
        atomicMaxFloat(&mmax[dst * N_HEAD + lane], v);
    }
}

// e_exp = exp(e - m[dst]); segment-sum into denom
__global__ void edge_exp_kernel(const int* __restrict__ edge_dst, const float* __restrict__ mmax,
                                float* __restrict__ ebuf, float* __restrict__ denom)
{
    long i = (long)blockIdx.x * blockDim.x + threadIdx.x;
    if (i >= (long)NEDGE * N_HEAD) return;
    int e = (int)(i >> 3), h = (int)(i & 7);
    int dst = edge_dst[e];
    float v = expf(ebuf[i] - mmax[dst * N_HEAD + h]);
    ebuf[i] = v;
    atomicAdd(&denom[dst * N_HEAD + h], v);
}

// h[dst] += alpha * z[src]; wave per edge
__global__ __launch_bounds__(256)
void aggregate_kernel(const int* __restrict__ edge_src, const int* __restrict__ edge_dst,
                      const float* __restrict__ ebuf, const float* __restrict__ denom,
                      const float* __restrict__ z, float* __restrict__ hacc)
{
    int wave = threadIdx.x >> 5, lane = threadIdx.x & 31;
    long e = (long)blockIdx.x * 8 + wave;
    if (e >= NEDGE) return;
    int src = edge_src[e], dst = edge_dst[e];
    float alpha_own = 0.0f;
    if (lane < N_HEAD)
        alpha_own = ebuf[e * N_HEAD + lane] / fmaxf(denom[dst * N_HEAD + lane], 1e-30f);
    const float* zr = z + (long)src * DMODEL;
    float* hr = hacc + (long)dst * DMODEL;
#pragma unroll
    for (int it = 0; it < 16; ++it) {
        float al = __shfl(alpha_own, it >> 1, 32);
        int c = it * 32 + lane;
        atomicAdd(&hr[c], al * zr[c]);
    }
}

// elu + residual; emit fp32 (residual path) and packed bf16 (GEMM A path)
__global__ void elu_residual_kernel(const float* __restrict__ hacc,
                                    const float* __restrict__ tgt_h,
                                    float* __restrict__ hres,
                                    unsigned int* __restrict__ hres_bfU)
{
    long i = (long)blockIdx.x * blockDim.x + threadIdx.x;
    if (i >= (long)N_TGT * DMODEL / 2) return;
    float2 x = *(const float2*)&hacc[i * 2];
    float2 t = *(const float2*)&tgt_h[i * 2];
    float y0 = ((x.x > 0.0f) ? x.x : (expf(x.x) - 1.0f)) + t.x;
    float y1 = ((x.y > 0.0f) ? x.y : (expf(x.y) - 1.0f)) + t.y;
    float2 y; y.x = y0; y.y = y1;
    *(float2*)&hres[i * 2] = y;
    hres_bfU[i] = pack_bf16(y0, y1);
}

__global__ void init_kernel(float* __restrict__ mmax, float* __restrict__ denom,
                            float* __restrict__ hacc)
{
    long i = (long)blockIdx.x * blockDim.x + threadIdx.x;
    if (i < (long)N_TGT * DMODEL) hacc[i] = 0.0f;
    if (i < (long)N_TGT * N_HEAD) { mmax[i] = -__builtin_inff(); denom[i] = 0.0f; }
}

// LayerNorm over 512-wide rows, wave per row, in place on d_out
__global__ __launch_bounds__(256)
void layernorm_kernel(float* __restrict__ out, const float* __restrict__ g,
                      const float* __restrict__ b)
{
    int wave = threadIdx.x >> 5, lane = threadIdx.x & 31;
    int row = blockIdx.x * 8 + wave;
    if (row >= N_TGT) return;
    float* o = out + (long)row * DMODEL;
    float x[16];
    float s = 0.0f, s2 = 0.0f;
#pragma unroll
    for (int i = 0; i < 16; ++i) {
        x[i] = o[lane + i * 32];
        s += x[i]; s2 += x[i] * x[i];
    }
    s = warp_sum(s); s2 = warp_sum(s2);
    float mu  = s * (1.0f / DMODEL);
    float var = s2 * (1.0f / DMODEL) - mu * mu;
    float rstd = rsqrtf(var + LN_EPS);
#pragma unroll
    for (int i = 0; i < 16; ++i) {
        int c = lane + i * 32;
        o[c] = (x[i] - mu) * rstd * g[c] + b[c];
    }
}

extern "C" void kernel_launch(void* const* d_in, const int* in_sizes, int n_in,
                              void* d_out, int out_size, void* d_ws, size_t ws_size,
                              hipStream_t stream)
{
    (void)in_sizes; (void)n_in; (void)out_size; (void)ws_size;
    const float* src_h      = (const float*)d_in[0];
    const float* tgt_h      = (const float*)d_in[1];
    const float* edge_embed = (const float*)d_in[2];
    const int*   edge_src   = (const int*)d_in[3];
    const int*   edge_dst   = (const int*)d_in[4];
    const float* W_fc       = (const float*)d_in[5];
    const float* W_feat     = (const float*)d_in[6];
    const float* attn_a     = (const float*)d_in[7];
    const float* w1         = (const float*)d_in[8];
    const float* b1         = (const float*)d_in[9];
    const float* w2         = (const float*)d_in[10];
    const float* b2         = (const float*)d_in[11];
    const float* ln_g       = (const float*)d_in[12];
    const float* ln_b       = (const float*)d_in[13];
    float* out = (float*)d_out;

    // workspace carve-up (~133 MB total)
    char* ws = (char*)d_ws;
    size_t off = 0;
    auto carve = [&](size_t bytes) {
        void* p = ws + off;
        off = (off + bytes + 255) & ~(size_t)255;
        return p;
    };
    float* z        = (float*)carve((size_t)N_SRC * DMODEL * 4);
    float* hacc     = (float*)carve((size_t)N_TGT * DMODEL * 4);
    float* hres     = (float*)carve((size_t)N_TGT * DMODEL * 4);
    unsigned int* src_bfU  = (unsigned int*)carve((size_t)N_SRC * DMODEL * 2);
    unsigned int* hres_bfU = (unsigned int*)carve((size_t)N_TGT * DMODEL * 2);
    __bf16* act_bf  = (__bf16*)carve((size_t)N_TGT * FFN_HID * 2);
    unsigned int* wfc_tU = (unsigned int*)carve((size_t)IN_DIM * DMODEL * 2);
    unsigned int* w1_tU  = (unsigned int*)carve((size_t)DMODEL * FFN_HID * 2);
    unsigned int* w2_tU  = (unsigned int*)carve((size_t)FFN_HID * DMODEL * 2);
    float* s_src  = (float*)carve((size_t)N_SRC * N_HEAD * 4);
    float* mmax   = (float*)carve((size_t)N_TGT * N_HEAD * 4);
    float* denom  = (float*)carve((size_t)N_TGT * N_HEAD * 4);
    float* ebuf   = (float*)carve((size_t)NEDGE * N_HEAD * 4);
    float* Cmat   = (float*)carve((size_t)EDGE_DIM * N_HEAD * 4);

    // --- pre-pack operands to bf16 ---
    {
        long np = (long)N_SRC * DMODEL / 2;
        pack_bf16_kernel<<<(int)((np + 255) / 256), 256, 0, stream>>>(src_h, src_bfU, np);
    }
    transpose_pack_kernel<<<dim3(IN_DIM / 32, DMODEL / 32), 256, 0, stream>>>(W_fc, wfc_tU, IN_DIM, DMODEL);
    transpose_pack_kernel<<<dim3(DMODEL / 32, FFN_HID / 32), 256, 0, stream>>>(w1, w1_tU, DMODEL, FFN_HID);
    transpose_pack_kernel<<<dim3(FFN_HID / 32, DMODEL / 32), 256, 0, stream>>>(w2, w2_tU, FFN_HID, DMODEL);

    {   // init accumulators (m = -inf, denom = 0, hacc = 0)
        long n = (long)N_TGT * DMODEL;
        init_kernel<<<(int)((n + 255) / 256), 256, 0, stream>>>(mmax, denom, hacc);
    }
    // z = src_h @ W_fc  (fp32 out)
    gemm_bf16_wmma<0, false><<<dim3((N_SRC + BM - 1) / BM, DMODEL / BN), 256, 0, stream>>>(
        (const __bf16*)src_bfU, (const __bf16*)wfc_tU, z, nullptr, nullptr, N_SRC, DMODEL, IN_DIM);
    compute_C_kernel<<<1, 128, 0, stream>>>(W_feat, attn_a, Cmat);
    s_src_kernel<<<N_SRC, 256, 0, stream>>>(z, attn_a, s_src);
    edge_score_kernel<<<(NEDGE + 7) / 8, 256, 0, stream>>>(
        edge_embed, edge_src, edge_dst, Cmat, s_src, ebuf, mmax);
    {
        long n = (long)NEDGE * N_HEAD;
        edge_exp_kernel<<<(int)((n + 255) / 256), 256, 0, stream>>>(edge_dst, mmax, ebuf, denom);
    }
    aggregate_kernel<<<(NEDGE + 7) / 8, 256, 0, stream>>>(
        edge_src, edge_dst, ebuf, denom, z, hacc);
    {
        long n = (long)N_TGT * DMODEL / 2;
        elu_residual_kernel<<<(int)((n + 255) / 256), 256, 0, stream>>>(hacc, tgt_h, hres, hres_bfU);
    }
    // FFN layer 1: act = relu(hres @ w1 + b1)  -> bf16 out
    gemm_bf16_wmma<1, true><<<dim3((N_TGT + BM - 1) / BM, FFN_HID / BN), 256, 0, stream>>>(
        (const __bf16*)hres_bfU, (const __bf16*)w1_tU, act_bf, b1, nullptr, N_TGT, FFN_HID, DMODEL);
    // FFN layer 2 + residual: out = act @ w2 + b2 + hres  (fp32 out)
    gemm_bf16_wmma<2, false><<<dim3((N_TGT + BM - 1) / BM, DMODEL / BN), 256, 0, stream>>>(
        act_bf, (const __bf16*)w2_tU, out, b2, hres, N_TGT, DMODEL, FFN_HID);
    layernorm_kernel<<<(N_TGT + 7) / 8, 256, 0, stream>>>(out, ln_g, ln_b);
}
